// BertTransformerWSD_84172769067860
// MI455X (gfx1250) — compile-verified, matrix operand order checked by reading, not run on hardware
//
#include <hip/hip_runtime.h>
#include <hip/hip_bf16.h>
#include <math.h>

typedef __bf16 bf16_t;
typedef __attribute__((ext_vector_type(16))) __bf16 v16bf;
typedef __attribute__((ext_vector_type(8)))  float   v8f;
typedef __attribute__((ext_vector_type(4)))  int     v4i;

#if __has_builtin(__builtin_amdgcn_global_load_async_to_lds_b128) && \
    __has_builtin(__builtin_amdgcn_s_wait_asynccnt)
#define HAVE_ASYNC_LDS 1
#else
#define HAVE_ASYNC_LDS 0
#endif

#define B_    16
#define S_    512
#define T_    256
#define DBERT 768
#define POSD  32
#define ENC   800
#define DM    512
#define H_    8
#define DH    64
#define L_    4
#define FF    2048
#define SENSES 30000
#define MTOK  (B_ * T_)          // 4096 token rows

#define BM 128                   // block tile M (4 waves x 32 rows)
#define BN 64                    // block tile N
#define BK 32                    // K step (one bf16 WMMA K)
#define LDP (BK + 8)             // padded LDS row: 80 B = 20 dwords -> conflict-free

union frag_u { v16bf v; uint4 q[2]; };

// ---------------------------------------------------------------------------
// Generic batched bf16 GEMM:  C = f(A[M,K] @ B[K,N])
//   bias != null : C = alpha * (AB + bias[col])              (linear layers)
//   lens != null : C = alpha * AB + (col>=lens[z/heads]?-1e9:0)  (attn scores)
//   else         : C = alpha * AB
// Block = 128 threads (4 waves). Each wave owns a 32-row strip: 2 A frags x
// 4 B frags -> 8 V_WMMA_F32_16X16X32_BF16 per K-step (12 ds_load_b128).
// A tile staged with GLOBAL_LOAD_ASYNC_TO_LDS_B128 (ASYNCcnt path, no VGPR
// round-trip); B tile staged TRANSPOSED (BsT[n][k]) so both fragment gathers
// are contiguous 16-B ds_load_b128 per CDNA5 ISA 7.12.2 layouts.
// All M used are multiples of BM, so A staging is branch-free (row clamp).
// ---------------------------------------------------------------------------
__global__ __launch_bounds__(128)
void gemm_bf16_wmma(const bf16_t* __restrict__ Ag, const bf16_t* __restrict__ Bg,
                    float* __restrict__ Cg,
                    int M, int N, int K,
                    long long sA, long long sB, long long sC,
                    const float* __restrict__ bias,
                    const int* __restrict__ lens, int heads,
                    float alpha)
{
    __shared__ __align__(16) bf16_t As [BM][LDP];   // As[m][k]
    __shared__ __align__(16) bf16_t BsT[BN][LDP];   // BsT[n][k]  (transposed tile)

    const int batch = blockIdx.z;
    const bf16_t* A  = Ag + (long long)batch * sA;
    const bf16_t* Bm = Bg + (long long)batch * sB;
    float*        C  = Cg + (long long)batch * sC;

    const int n0 = blockIdx.x * BN;
    const int m0 = blockIdx.y * BM;

    const int tid  = threadIdx.x;
    const int lane = tid & 31;
    const int wave = tid >> 5;        // 0..3
    const int hlf  = lane >> 4;       // lane half selects K-subranges
    const int l16  = lane & 15;

    v8f acc[2][4] = {};

    const int br = tid >> 2;          // B tile: 32 k-rows, 4 thr/row x 16 cols
    const int bc = (tid & 3) * 16;

    // A row for this thread (clamped; all M in this model are multiples of BM)
    const int arow = (m0 + tid < M) ? (m0 + tid) : (M - 1);

    for (int k0 = 0; k0 < K; k0 += BK) {
        // ---- stage A tile: 128 rows x 32 k, one full row (64 B) per thread
        {
            const bf16_t* src = A + (size_t)arow * K + k0;
#if HAVE_ASYNC_LDS
            // memory -> LDS direct, tracked by ASYNCcnt (no VGPR round-trip)
            #pragma unroll
            for (int i = 0; i < 4; ++i) {
                __builtin_amdgcn_global_load_async_to_lds_b128(
                    (__attribute__((address_space(1))) v4i*)(src + i * 8),
                    (__attribute__((address_space(3))) v4i*)&As[tid][i * 8],
                    0, 0);
            }
#else
            const uint4* s4 = (const uint4*)src;
            uint4 r0 = s4[0], r1 = s4[1], r2 = s4[2], r3 = s4[3];
            *(uint4*)&As[tid][0]  = r0;
            *(uint4*)&As[tid][8]  = r1;
            *(uint4*)&As[tid][16] = r2;
            *(uint4*)&As[tid][24] = r3;
#endif
        }
        // ---- stage B tile transposed (N-edge guarded: vocab GEMM N=30000)
        {
            const int gcol = n0 + bc;
            const size_t rowoff = (size_t)(k0 + br) * N;
            bf16_t regs[16];
            if (gcol + 16 <= N) {
                const uint4* src = (const uint4*)(Bm + rowoff + gcol);
                uint4 r0 = src[0], r1 = src[1];
                *(uint4*)&regs[0] = r0;
                *(uint4*)&regs[8] = r1;
            } else {
                #pragma unroll
                for (int i = 0; i < 16; ++i) {
                    const int c = gcol + i;
                    regs[i] = (c < N) ? Bm[rowoff + c] : (bf16_t)0.f;
                }
            }
            #pragma unroll
            for (int i = 0; i < 16; ++i) BsT[bc + i][br] = regs[i];
        }
#if HAVE_ASYNC_LDS
        __builtin_amdgcn_s_wait_asynccnt(0);   // A-tile LDS writes landed
#endif
        __syncthreads();

        if (k0 + BK < K) {   // global_prefetch_b8 next K-tile
            __builtin_prefetch(A  + (size_t)arow * K + (k0 + BK), 0, 1);
            __builtin_prefetch(Bm + (size_t)(k0 + BK + br) * N + (n0 + bc), 0, 1);
        }

        // ---- A fragments: lanes 0-15 hold K{0..7,16..23}, lanes 16-31 K{8..15,24..31}
        const int koff = hlf * 8;
        const int mbase = wave * 32 + l16;
        frag_u a0, a1;
        a0.q[0] = *(const uint4*)&As[mbase][koff];
        a0.q[1] = *(const uint4*)&As[mbase][16 + koff];
        a1.q[0] = *(const uint4*)&As[mbase + 16][koff];
        a1.q[1] = *(const uint4*)&As[mbase + 16][16 + koff];

        // ---- B fragments (same K striping, N = lane%16); each feeds 2 WMMAs
        //      uniform control flow: EXEC all ones at every wmma
        #pragma unroll
        for (int j = 0; j < 4; ++j) {
            const int c = j * 16 + l16;
            frag_u bfr;
            bfr.q[0] = *(const uint4*)&BsT[c][koff];
            bfr.q[1] = *(const uint4*)&BsT[c][16 + koff];
            acc[0][j] = __builtin_amdgcn_wmma_f32_16x16x32_bf16(
                            false, a0.v, false, bfr.v, (short)0, acc[0][j], false, false);
            acc[1][j] = __builtin_amdgcn_wmma_f32_16x16x32_bf16(
                            false, a1.v, false, bfr.v, (short)0, acc[1][j], false, false);
        }
        __syncthreads();
    }

    // ---- epilogue: D layout = lanes 0-15 M=r, lanes 16-31 M=r+8, N=lane%16
    #pragma unroll
    for (int j = 0; j < 4; ++j) {
        const int col = n0 + j * 16 + l16;
        if (col >= N) continue;
        const float badd = bias ? bias[col] : 0.f;
        const float mask = lens ? ((col >= lens[batch / heads]) ? -1e9f : 0.f) : 0.f;
        #pragma unroll
        for (int mi = 0; mi < 2; ++mi) {
            #pragma unroll
            for (int r = 0; r < 8; ++r) {
                const int row = m0 + wave * 32 + mi * 16 + hlf * 8 + r;
                if (row < M) {
                    float o = acc[mi][j][r];
                    o = bias ? alpha * (o + badd) : (alpha * o + mask);
                    C[(size_t)row * N + col] = o;
                }
            }
        }
    }
}

// ---------------------------------------------------------------------------
// Segment mean-pool + POS-embed concat -> bf16 feature rows [4096, 800].
// seg_ids per batch are non-decreasing, so each word's wordpieces are a
// contiguous slice: two binary searches instead of an O(S) scan.
// ---------------------------------------------------------------------------
__global__ __launch_bounds__(128)
void pool_concat(const float* __restrict__ x, const int* __restrict__ seg_ids,
                 const int* __restrict__ pos_tags, const float* __restrict__ pos_table,
                 bf16_t* __restrict__ feat)
{
    const int bt = blockIdx.x;
    const int b = bt / T_, t = bt % T_;
    const int* seg = seg_ids + b * S_;
    int lo = 0, hi = S_;
    while (lo < hi) { int mid = (lo + hi) >> 1; if (seg[mid] < t) lo = mid + 1; else hi = mid; }
    const int s0 = lo;
    hi = S_;
    while (lo < hi) { int mid = (lo + hi) >> 1; if (seg[mid] < t + 1) lo = mid + 1; else hi = mid; }
    const int s1 = lo;
    const float invc = 1.f / fmaxf((float)(s1 - s0), 1.f);
    const int tag = pos_tags[bt];
    bf16_t* frow = feat + (size_t)bt * ENC;
    for (int d = threadIdx.x; d < ENC; d += 128) {
        float v;
        if (d < DBERT) {
            float s = 0.f;
            for (int si = s0; si < s1; ++si) s += x[((size_t)b * S_ + si) * DBERT + d];
            v = s * invc;
        } else {
            v = pos_table[tag * POSD + (d - DBERT)];
        }
        frow[d] = (bf16_t)v;
    }
}

__global__ void f32_to_bf16(const float* __restrict__ in, bf16_t* __restrict__ out, long long n)
{
    long long i = (long long)blockIdx.x * blockDim.x + threadIdx.x;
    const long long stride = (long long)gridDim.x * blockDim.x;
    for (; i < n; i += stride) out[i] = (bf16_t)in[i];
}

__global__ void relu_f32_to_bf16(const float* __restrict__ in, bf16_t* __restrict__ out, long long n)
{
    long long i = (long long)blockIdx.x * blockDim.x + threadIdx.x;
    const long long stride = (long long)gridDim.x * blockDim.x;
    for (; i < n; i += stride) out[i] = (bf16_t)fmaxf(in[i], 0.f);
}

// qkv [4096,1536] f32 -> q[b,h,t,d], kT[b,h,d,t], v[b,h,t,d] bf16
__global__ void split_qkv(const float* __restrict__ qkv,
                          bf16_t* __restrict__ q, bf16_t* __restrict__ kT, bf16_t* __restrict__ v)
{
    long long idx = (long long)blockIdx.x * blockDim.x + threadIdx.x;
    const long long total  = (long long)MTOK * (3 * DM);
    const long long stride = (long long)gridDim.x * blockDim.x;
    for (; idx < total; idx += stride) {
        const int row = (int)(idx / (3 * DM));
        const int col = (int)(idx % (3 * DM));
        const int b = row / T_, t = row % T_;
        const int which = col / DM;
        const int c = col % DM;
        const int h = c / DH, d = c % DH;
        const bf16_t val = (bf16_t)qkv[idx];
        const int bh = b * H_ + h;
        if (which == 0)      q [((size_t)bh * T_ + t) * DH + d] = val;
        else if (which == 1) kT[((size_t)bh * DH + d) * T_ + t] = val;
        else                 v [((size_t)bh * T_ + t) * DH + d] = val;
    }
}

// ctx [b,h,t,d] f32 -> [b*t, h*64+d] bf16
__global__ void merge_ctx(const float* __restrict__ ctx, bf16_t* __restrict__ out)
{
    long long idx = (long long)blockIdx.x * blockDim.x + threadIdx.x;
    const long long total  = (long long)MTOK * DM;
    const long long stride = (long long)gridDim.x * blockDim.x;
    for (; idx < total; idx += stride) {
        const int row = (int)(idx / DM), c = (int)(idx % DM);
        const int b = row / T_, t = row % T_, h = c / DH, d = c % DH;
        out[idx] = (bf16_t)ctx[(((size_t)(b * H_ + h)) * T_ + t) * DH + d];
    }
}

__global__ __launch_bounds__(256)
void softmax_rows(const float* __restrict__ Sc, bf16_t* __restrict__ P)
{
    __shared__ float red[256];
    const size_t row = blockIdx.x;
    const int tid = threadIdx.x;
    const float v = Sc[row * T_ + tid];
    red[tid] = v; __syncthreads();
    for (int s = 128; s > 0; s >>= 1) { if (tid < s) red[tid] = fmaxf(red[tid], red[tid + s]); __syncthreads(); }
    const float mx = red[0]; __syncthreads();
    const float e = __expf(v - mx);
    red[tid] = e; __syncthreads();
    for (int s = 128; s > 0; s >>= 1) { if (tid < s) red[tid] += red[tid + s]; __syncthreads(); }
    P[row * T_ + tid] = (bf16_t)(e / red[0]);
}

// h = LN(h + delta); also emit bf16 copy for the next GEMM input.
__global__ __launch_bounds__(256)
void residual_layernorm(float* __restrict__ h, const float* __restrict__ delta,
                        const float* __restrict__ g, const float* __restrict__ b,
                        bf16_t* __restrict__ hb)
{
    __shared__ float red[256];
    const size_t row = blockIdx.x;
    const int tid = threadIdx.x;
    const float x0 = h[row * DM + tid]       + delta[row * DM + tid];
    const float x1 = h[row * DM + tid + 256] + delta[row * DM + tid + 256];
    red[tid] = x0 + x1; __syncthreads();
    for (int s = 128; s > 0; s >>= 1) { if (tid < s) red[tid] += red[tid + s]; __syncthreads(); }
    const float mu = red[0] * (1.f / DM); __syncthreads();
    const float d0 = x0 - mu, d1 = x1 - mu;
    red[tid] = d0 * d0 + d1 * d1; __syncthreads();
    for (int s = 128; s > 0; s >>= 1) { if (tid < s) red[tid] += red[tid + s]; __syncthreads(); }
    const float rstd = rsqrtf(red[0] * (1.f / DM) + 1e-5f);
    const float y0 = d0 * rstd * g[tid]       + b[tid];
    const float y1 = d1 * rstd * g[tid + 256] + b[tid + 256];
    h[row * DM + tid]        = y0;  h[row * DM + tid + 256]  = y1;
    hb[row * DM + tid]       = (bf16_t)y0;
    hb[row * DM + tid + 256] = (bf16_t)y1;
}

// in-place log_softmax over rows of 30000
__global__ __launch_bounds__(256)
void log_softmax_rows(float* __restrict__ logits)
{
    __shared__ float red[256];
    float* p = logits + (size_t)blockIdx.x * SENSES;
    const int tid = threadIdx.x;
    float mx = -1e30f;
    for (int c = tid; c < SENSES; c += 256) mx = fmaxf(mx, p[c]);
    red[tid] = mx; __syncthreads();
    for (int s = 128; s > 0; s >>= 1) { if (tid < s) red[tid] = fmaxf(red[tid], red[tid + s]); __syncthreads(); }
    mx = red[0]; __syncthreads();
    float sum = 0.f;
    for (int c = tid; c < SENSES; c += 256) sum += __expf(p[c] - mx);
    red[tid] = sum; __syncthreads();
    for (int s = 128; s > 0; s >>= 1) { if (tid < s) red[tid] += red[tid + s]; __syncthreads(); }
    const float lse = mx + __logf(red[0]);
    for (int c = tid; c < SENSES; c += 256) p[c] = p[c] - lse;
}

// ---------------------------------------------------------------------------
extern "C" void kernel_launch(void* const* d_in, const int* in_sizes, int n_in,
                              void* d_out, int out_size, void* d_ws, size_t ws_size,
                              hipStream_t stream)
{
    (void)in_sizes; (void)n_in; (void)out_size; (void)ws_size;
    const float* x        = (const float*)d_in[0];
    const int*   seg_ids  = (const int*)  d_in[1];
    const int*   text_len = (const int*)  d_in[2];
    const int*   pos_tags = (const int*)  d_in[3];
    const float* pos_tab  = (const float*)d_in[4];
    const float* Wp       = (const float*)d_in[5];
    const float* bp       = (const float*)d_in[6];
    const float* Wqkv     = (const float*)d_in[7];
    const float* bqkv     = (const float*)d_in[8];
    const float* Wo       = (const float*)d_in[9];
    const float* bo       = (const float*)d_in[10];
    const float* ln1g     = (const float*)d_in[11];
    const float* ln1b     = (const float*)d_in[12];
    const float* W1       = (const float*)d_in[13];
    const float* b1       = (const float*)d_in[14];
    const float* W2       = (const float*)d_in[15];
    const float* b2       = (const float*)d_in[16];
    const float* ln2g     = (const float*)d_in[17];
    const float* ln2b     = (const float*)d_in[18];
    const float* Wout     = (const float*)d_in[19];
    const float* bout     = (const float*)d_in[20];

    char* ws = (char*)d_ws;
    size_t off = 0;
    auto alloc = [&](size_t bytes) -> void* {
        void* p = ws + off;
        off += (bytes + 255) & ~(size_t)255;
        return p;
    };

    bf16_t* featB  = (bf16_t*)alloc((size_t)MTOK * ENC * 2);
    bf16_t* WpB    = (bf16_t*)alloc((size_t)ENC * DM * 2);
    bf16_t* WqkvB  = (bf16_t*)alloc((size_t)L_ * DM * 3 * DM * 2);
    bf16_t* WoB    = (bf16_t*)alloc((size_t)L_ * DM * DM * 2);
    bf16_t* W1B    = (bf16_t*)alloc((size_t)L_ * DM * FF * 2);
    bf16_t* W2B    = (bf16_t*)alloc((size_t)L_ * FF * DM * 2);
    bf16_t* WoutB  = (bf16_t*)alloc((size_t)DM * SENSES * 2);
    float*  h32    = (float*) alloc((size_t)MTOK * DM * 4);
    bf16_t* hB     = (bf16_t*)alloc((size_t)MTOK * DM * 2);
    float*  qkv32  = (float*) alloc((size_t)MTOK * 3 * DM * 4);
    bf16_t* qB     = (bf16_t*)alloc((size_t)B_ * H_ * T_ * DH * 2);
    bf16_t* kTB    = (bf16_t*)alloc((size_t)B_ * H_ * DH * T_ * 2);
    bf16_t* vB     = (bf16_t*)alloc((size_t)B_ * H_ * T_ * DH * 2);
    float*  sc32   = (float*) alloc((size_t)B_ * H_ * T_ * T_ * 4);
    bf16_t* PB     = (bf16_t*)alloc((size_t)B_ * H_ * T_ * T_ * 2);
    float*  ctx32  = (float*) alloc((size_t)B_ * H_ * T_ * DH * 4);
    bf16_t* ctxB   = (bf16_t*)alloc((size_t)MTOK * DM * 2);
    float*  tmp32  = (float*) alloc((size_t)MTOK * DM * 4);
    float*  ff132  = (float*) alloc((size_t)MTOK * FF * 4);
    bf16_t* ff1B   = (bf16_t*)alloc((size_t)MTOK * FF * 2);

    auto conv = [&](const float* src, bf16_t* dst, long long n) {
        f32_to_bf16<<<2048, 256, 0, stream>>>(src, dst, n);
    };
    auto gemm = [&](const bf16_t* A, const bf16_t* Bm, float* C,
                    int M, int N, int K,
                    long long sA, long long sB, long long sC, int batch,
                    const float* bias, const int* lens, int heads, float alpha) {
        dim3 g((N + BN - 1) / BN, (M + BM - 1) / BM, batch);
        gemm_bf16_wmma<<<g, 128, 0, stream>>>(A, Bm, C, M, N, K, sA, sB, sC,
                                              bias, lens, heads, alpha);
    };

    // weight conversion (L2-resident thereafter: <60 MB total vs 192 MB L2)
    conv(Wp,   WpB,   (long long)ENC * DM);
    conv(Wqkv, WqkvB, (long long)L_ * DM * 3 * DM);
    conv(Wo,   WoB,   (long long)L_ * DM * DM);
    conv(W1,   W1B,   (long long)L_ * DM * FF);
    conv(W2,   W2B,   (long long)L_ * FF * DM);
    conv(Wout, WoutB, (long long)DM * SENSES);

    // pooling + concat -> feat bf16
    pool_concat<<<MTOK, 128, 0, stream>>>(x, seg_ids, pos_tags, pos_tab, featB);

    // h = (feat @ Wp + bp) * sqrt(800)
    const float SCALE = 28.2842712474619f;
    gemm(featB, WpB, h32, MTOK, DM, ENC, 0, 0, 0, 1, bp, nullptr, 1, SCALE);
    conv(h32, hB, (long long)MTOK * DM);

    for (int l = 0; l < L_; ++l) {
        gemm(hB, WqkvB + (size_t)l * DM * 3 * DM, qkv32, MTOK, 3 * DM, DM,
             0, 0, 0, 1, bqkv + (size_t)l * 3 * DM, nullptr, 1, 1.f);
        split_qkv<<<2048, 256, 0, stream>>>(qkv32, qB, kTB, vB);

        // scores = q @ kT / 8 + pad-mask bias   (batched over B*H)
        gemm(qB, kTB, sc32, T_, T_, DH,
             (long long)T_ * DH, (long long)DH * T_, (long long)T_ * T_, B_ * H_,
             nullptr, text_len, H_, 0.125f);
        softmax_rows<<<B_ * H_ * T_, 256, 0, stream>>>(sc32, PB);

        // ctx = P @ v
        gemm(PB, vB, ctx32, T_, DH, T_,
             (long long)T_ * T_, (long long)T_ * DH, (long long)T_ * DH, B_ * H_,
             nullptr, nullptr, 1, 1.f);
        merge_ctx<<<2048, 256, 0, stream>>>(ctx32, ctxB);

        gemm(ctxB, WoB + (size_t)l * DM * DM, tmp32, MTOK, DM, DM,
             0, 0, 0, 1, bo + (size_t)l * DM, nullptr, 1, 1.f);
        residual_layernorm<<<MTOK, 256, 0, stream>>>(h32, tmp32,
                                                     ln1g + (size_t)l * DM,
                                                     ln1b + (size_t)l * DM, hB);

        gemm(hB, W1B + (size_t)l * DM * FF, ff132, MTOK, FF, DM,
             0, 0, 0, 1, b1 + (size_t)l * FF, nullptr, 1, 1.f);
        relu_f32_to_bf16<<<4096, 256, 0, stream>>>(ff132, ff1B, (long long)MTOK * FF);
        gemm(ff1B, W2B + (size_t)l * FF * DM, tmp32, MTOK, DM, FF,
             0, 0, 0, 1, b2 + (size_t)l * DM, nullptr, 1, 1.f);
        residual_layernorm<<<MTOK, 256, 0, stream>>>(h32, tmp32,
                                                     ln2g + (size_t)l * DM,
                                                     ln2b + (size_t)l * DM, hB);
    }

    // vocab projection straight into d_out, then in-place log_softmax
    float* logits = (float*)d_out;
    gemm(hB, WoutB, logits, MTOK, SENSES, DM, 0, 0, 0, 1, bout, nullptr, 1, 1.f);
    log_softmax_rows<<<MTOK, 256, 0, stream>>>(logits);
}
